// QADebiasedContrastiveLoss_34789235097603
// MI455X (gfx1250) — matrix operand order repaired and checked by reading, not compiled, not run on hardware
//
#include <hip/hip_runtime.h>
#include <hip/hip_bf16.h>

// ---------------------------------------------------------------------------
// QA debiased contrastive loss, CDNA5 (gfx1250) WMMA implementation.
// S=128 sections, Q=64 questions, P=64 sentences, D=256.
// Big GEMM (8192x8192x256) + small GEMM (128x8192x256) via
// v_wmma_f32_16x16x32_bf16 (f32 accumulate), exp/log epilogues in VALU.
// Sentence panels stream into LDS via GLOBAL_LOAD_ASYNC_TO_LDS_B128
// (ASYNCcnt) double-buffered against WMMA compute when available.
// ---------------------------------------------------------------------------

typedef __bf16 bf16;
typedef __attribute__((ext_vector_type(16))) __bf16 v16bf;
typedef __attribute__((ext_vector_type(8)))  __bf16 v8bf;
typedef __attribute__((ext_vector_type(8)))  float  v8f;

union FragA { v16bf v; v8bf h[2]; };

#define NS   128      // sections
#define NQ   64       // questions / section
#define NP   64       // sentences / section
#define ND   256      // embedding dim
#define NROWS (NS*NQ) // 8192
#define INV_T (1.0f/0.07f)

#if __has_builtin(__builtin_amdgcn_global_load_async_to_lds_b128) && \
    __has_builtin(__builtin_amdgcn_s_wait_asynccnt)
#define HAVE_ASYNC_LDS 1
// builtin signature (from clang diagnostic): param0 = AS(1) v4i32*, so mirror it
typedef int v4i_gcc __attribute__((vector_size(16)));
typedef __attribute__((address_space(1))) v4i_gcc* g4ptr;
typedef __attribute__((address_space(3))) v4i_gcc* l4ptr;
#else
#define HAVE_ASYNC_LDS 0
#endif

// issue a 32 KB panel copy (global -> LDS), 16 B per thread per step
__device__ __forceinline__ void copy_panel(const bf16* __restrict__ gsrc,
                                           bf16* __restrict__ ldst, int tid) {
#if HAVE_ASYNC_LDS
    const uint4* g = (const uint4*)gsrc;
    uint4*       l = (uint4*)ldst;
    #pragma unroll
    for (int i = tid; i < 2048; i += 256) {
        __builtin_amdgcn_global_load_async_to_lds_b128(
            (g4ptr)(g + i), (l4ptr)(l + i), 0, 0);
    }
#else
    const uint4* g = (const uint4*)gsrc;
    uint4*       l = (uint4*)ldst;
    #pragma unroll
    for (int i = tid; i < 2048; i += 256) l[i] = g[i];
#endif
}

__device__ __forceinline__ void panel_wait() {
#if HAVE_ASYNC_LDS
    __builtin_amdgcn_s_wait_asynccnt(0);
#endif
}

// ---------------------------------------------------------------------------
// zero the small f32 accumulator region (denom[128], numer[128], qs_sum[1])
__global__ void init_kernel(float* acc) {
    int t = threadIdx.x;
    if (t < 257) acc[t] = 0.0f;
}

// ---------------------------------------------------------------------------
// L2-normalize rows and convert to bf16.
// blocks: [0,8192) questions, [8192,16384) sentences, [16384,16512) titles
__global__ __launch_bounds__(256) void norm_kernel(
    const float* __restrict__ q, const float* __restrict__ sen,
    const float* __restrict__ st,
    bf16* __restrict__ qbf, bf16* __restrict__ sbf, bf16* __restrict__ stbf) {
    int b = blockIdx.x;
    const float* src; bf16* dst;
    if (b < NROWS)            { src = q   + (size_t)b * ND;             dst = qbf  + (size_t)b * ND; }
    else if (b < 2 * NROWS)   { src = sen + (size_t)(b - NROWS) * ND;   dst = sbf  + (size_t)(b - NROWS) * ND; }
    else                      { src = st  + (size_t)(b - 2*NROWS) * ND; dst = stbf + (size_t)(b - 2*NROWS) * ND; }

    int t = threadIdx.x;                 // 256 threads, one element each
    float x = src[t];
    float v = x * x;
    #pragma unroll
    for (int off = 16; off > 0; off >>= 1) v += __shfl_down(v, off, 32);
    __shared__ float red[8];
    int wave = t >> 5, lane = t & 31;
    if (lane == 0) red[wave] = v;
    __syncthreads();
    if (t == 0) {
        float s2 = 0.0f;
        #pragma unroll
        for (int i = 0; i < 8; ++i) s2 += red[i];
        red[0] = s2;
    }
    __syncthreads();
    float inv = 1.0f / fmaxf(sqrtf(red[0]), 1e-12f);
    dst[t] = (bf16)(x * inv);
}

// ---------------------------------------------------------------------------
// qs_loss: one block per section s. Rows = 64 questions of s,
// columns = all 8192 sentences, 128 panels of 64, double-buffered in LDS.
__global__ __launch_bounds__(256) void qs_kernel(
    const bf16* __restrict__ qbf, const bf16* __restrict__ sbf,
    float* __restrict__ qs_sum) {
    __shared__ char smem[65536];
    bf16* buf0 = (bf16*)smem;              // stage 1: A tile; then B double-buffer
    bf16* buf1 = (bf16*)(smem + 32768);

    const int tid   = threadIdx.x;
    const int s     = blockIdx.x;
    const int wave  = tid >> 5;
    const int lane  = tid & 31;
    const int mi    = wave & 3;            // 16-row M tile of this wave
    const int npair = wave >> 2;           // which pair of N tiles

    // ---- panel 0 (async) into buf1; A tile (sync) into buf0
    copy_panel(sbf, buf1, tid);
    {
        const uint4* src = (const uint4*)(qbf + (size_t)s * NQ * ND);
        uint4* dst = (uint4*)buf0;
        #pragma unroll
        for (int i = tid; i < 2048; i += 256) dst[i] = src[i];
    }
    __syncthreads();

    // ---- preload the 8 A fragments for this wave's row tile (contiguous k = 16*half + s order)
    FragA afrag[8];
    {
        int r = mi * 16 + (lane & 15);
        int khalf = lane >> 4;             // 0: K 0-15, 1: K 16-31
        const bf16* arow = buf0 + r * ND;
        #pragma unroll
        for (int kb = 0; kb < 8; ++kb) {
            afrag[kb].h[0] = *(const v8bf*)(arow + kb * 32 + khalf * 16);
            afrag[kb].h[1] = *(const v8bf*)(arow + kb * 32 + khalf * 16 + 8);
        }
    }
    panel_wait();        // panel 0 landed
    __syncthreads();     // all waves: afrag done (buf0 free), buf1 visible

    float acc8[8], ownAcc[8], posReg[2][8];
    #pragma unroll
    for (int i = 0; i < 8; ++i) {
        acc8[i] = 0.0f; ownAcc[i] = 0.0f; posReg[0][i] = 0.0f; posReg[1][i] = 0.0f;
    }

    for (int j = 0; j < NS; ++j) {
        bf16* cur = (j & 1) ? buf0 : buf1;     // panel j
        bf16* nxt = (j & 1) ? buf1 : buf0;     // panel j+1 target

        if (j + 1 < NS) {                      // overlap DMA with WMMA
            copy_panel(sbf + (size_t)(j + 1) * NP * ND, nxt, tid);
            if (j + 2 < NS)                    // L2 prefetch two panels ahead
                __builtin_prefetch(sbf + (size_t)(j + 2) * NP * ND + tid * 64, 0, 1);
        }

        #pragma unroll
        for (int t = 0; t < 2; ++t) {
            int nj = npair * 2 + t;
            v8f c = {};
            int col = nj * 16 + (lane & 15);
            const bf16* brow = cur + col * ND;
            int koff = (lane >> 4) * 16;       // B layout: lanes 0-15 K 0-15, 16-31 K 16-31
            #pragma unroll
            for (int kb = 0; kb < 8; ++kb) {
                v16bf bfrag = *(const v16bf*)(brow + kb * 32 + koff);
                c = __builtin_amdgcn_wmma_f32_16x16x32_bf16(
                        false, afrag[kb].v, false, bfrag, (short)0, c, false, false);
            }
            #pragma unroll
            for (int v = 0; v < 8; ++v) {
                float e = __expf(c[v] * INV_T);
                acc8[v] += e;
                if (j == s) { posReg[t][v] = e; ownAcc[v] += e; }
            }
        }

        panel_wait();                          // my async loads into nxt done
        __syncthreads();                       // everyone done reading cur / loading nxt
    }

    // ---- overlay reductions on the (now dead) LDS
    float* rowTot  = (float*)smem;             // 64
    float* rowOwn  = rowTot + 64;              // 64
    float* lossAcc = rowOwn + 64;              // 1
    if (tid < 129) ((float*)smem)[tid] = 0.0f;
    __syncthreads();

    const int rbase = mi * 16 + (lane >> 4) * 8;
    #pragma unroll
    for (int v = 0; v < 8; ++v) {
        atomicAdd(&rowTot[rbase + v], acc8[v]);
        atomicAdd(&rowOwn[rbase + v], ownAcc[v]);
    }
    __syncthreads();

    const float term2 = __expf(-INV_T) * (float)(NROWS - 1);
    float myloss = 0.0f;
    #pragma unroll
    for (int t = 0; t < 2; ++t) {
        #pragma unroll
        for (int v = 0; v < 8; ++v) {
            float pos = posReg[t][v];
            int   m   = rbase + v;
            float neg = rowTot[m] - rowOwn[m];
            float t1  = -0.1f * pos + neg * (1.0f / 0.9f);
            float Ng  = fmaxf(fmaxf(t1, term2), 1e-8f);
            myloss   += __logf((pos + Ng) / pos);   // == -log(pos/(pos+Ng))
        }
    }
    atomicAdd(lossAcc, myloss);
    __syncthreads();
    if (tid == 0) atomicAdd(qs_sum, lossAcc[0]);
}

// ---------------------------------------------------------------------------
// sq_loss GEMM: block t computes stn(128x256) x qn[t]^T(256x64) in two
// 64-row panels; exp row-sums go to denom[s] (and numer[t] for s==t).
__global__ __launch_bounds__(256) void sq_kernel(
    const bf16* __restrict__ stbf, const bf16* __restrict__ qbf,
    float* __restrict__ denom, float* __restrict__ numer) {
    __shared__ char smem[65536];
    bf16* Bbuf = (bf16*)smem;              // questions of section t (resident)
    bf16* Abuf = (bf16*)(smem + 32768);    // stn panel

    const int tid = threadIdx.x, tblk = blockIdx.x;
    const int wave = tid >> 5, lane = tid & 31;
    const int mi = wave & 3, npair = wave >> 2;

    {
        const uint4* src = (const uint4*)(qbf + (size_t)tblk * NQ * ND);
        #pragma unroll
        for (int i = tid; i < 2048; i += 256) ((uint4*)Bbuf)[i] = src[i];
    }

    for (int p = 0; p < 2; ++p) {
        __syncthreads();
        const uint4* srcA = (const uint4*)(stbf + (size_t)p * 64 * ND);
        #pragma unroll
        for (int i = tid; i < 2048; i += 256) ((uint4*)Abuf)[i] = srcA[i];
        __syncthreads();

        FragA af[8];
        int r = mi * 16 + (lane & 15), khalf = lane >> 4;
        const bf16* arow = Abuf + r * ND;
        #pragma unroll
        for (int kb = 0; kb < 8; ++kb) {
            af[kb].h[0] = *(const v8bf*)(arow + kb * 32 + khalf * 16);
            af[kb].h[1] = *(const v8bf*)(arow + kb * 32 + khalf * 16 + 8);
        }

        float racc[8];
        #pragma unroll
        for (int v = 0; v < 8; ++v) racc[v] = 0.0f;

        #pragma unroll
        for (int t = 0; t < 2; ++t) {
            int nj = npair * 2 + t;
            v8f c = {};
            int col = nj * 16 + (lane & 15);
            const bf16* brow = Bbuf + col * ND;
            int koff = (lane >> 4) * 16;
            #pragma unroll
            for (int kb = 0; kb < 8; ++kb) {
                v16bf bfrag = *(const v16bf*)(brow + kb * 32 + koff);
                c = __builtin_amdgcn_wmma_f32_16x16x32_bf16(
                        false, af[kb].v, false, bfrag, (short)0, c, false, false);
            }
            #pragma unroll
            for (int v = 0; v < 8; ++v) racc[v] += __expf(c[v] * INV_T);
        }

        // reduce across the 16 lanes that share the same 8 rows
        #pragma unroll
        for (int off = 1; off < 16; off <<= 1)
            #pragma unroll
            for (int v = 0; v < 8; ++v) racc[v] += __shfl_xor(racc[v], off, 32);

        if ((lane & 15) == 0) {
            int rbase = mi * 16 + (lane >> 4) * 8;
            #pragma unroll
            for (int v = 0; v < 8; ++v) {
                int srow = p * 64 + rbase + v;
                atomicAdd(&denom[srow], racc[v]);
                if (srow == tblk) atomicAdd(&numer[tblk], racc[v]);
            }
        }
    }
}

// ---------------------------------------------------------------------------
// out = qs_loss + sum_s(log(denom)-log(numer)) + at_loss (== -log(x/x) == 0)
__global__ void fin_kernel(const float* __restrict__ denom,
                           const float* __restrict__ numer,
                           const float* __restrict__ qs_sum,
                           float* __restrict__ out) {
    __shared__ float red[128];
    int t = threadIdx.x;                   // 128 threads
    red[t] = __logf(denom[t]) - __logf(numer[t]);
    __syncthreads();
    for (int off = 64; off > 0; off >>= 1) {
        if (t < off) red[t] += red[t + off];
        __syncthreads();
    }
    if (t == 0) out[0] = qs_sum[0] + red[0];   // at_loss is exactly 0
}

// ---------------------------------------------------------------------------
extern "C" void kernel_launch(void* const* d_in, const int* in_sizes, int n_in,
                              void* d_out, int out_size, void* d_ws, size_t ws_size,
                              hipStream_t stream) {
    (void)in_sizes; (void)n_in; (void)out_size; (void)ws_size;
    const float* at  = (const float*)d_in[0]; (void)at;  // at_loss == 0 exactly
    const float* st  = (const float*)d_in[1];
    const float* q   = (const float*)d_in[2];
    const float* sen = (const float*)d_in[3];

    char* ws = (char*)d_ws;
    bf16*  qbf    = (bf16*)ws;                       // 8192*256*2 = 4 MB
    bf16*  sbf    = (bf16*)(ws + 4194304);           // 4 MB
    bf16*  stbf   = (bf16*)(ws + 8388608);           // 64 KB
    float* denom  = (float*)(ws + 8454144);          // 128 f32
    float* numer  = denom + 128;                     // 128 f32
    float* qs_sum = numer + 128;                     // 1 f32

    init_kernel<<<1, 288, 0, stream>>>(denom);
    norm_kernel<<<2 * NROWS + NS, 256, 0, stream>>>(q, sen, st, qbf, sbf, stbf);
    qs_kernel<<<NS, 256, 0, stream>>>(qbf, sbf, qs_sum);
    sq_kernel<<<NS, 256, 0, stream>>>(stbf, qbf, denom, numer);
    fin_kernel<<<1, 128, 0, stream>>>(denom, numer, qs_sum, (float*)d_out);
}
